// SelfAttentionV3_51599737094691
// MI455X (gfx1250) — matrix operand order repaired and checked by reading, not compiled, not run on hardware
//
#include <hip/hip_runtime.h>

// ---------------------------------------------------------------------------
// Types for CDNA5 WMMA (wave32): v_wmma_f32_16x16x32_bf16
// ---------------------------------------------------------------------------
typedef __attribute__((ext_vector_type(16))) __bf16 v16bf;
typedef __attribute__((ext_vector_type(8)))  __bf16 v8bf;
typedef __attribute__((ext_vector_type(8)))  float  v8f;

static __device__ __forceinline__ unsigned short f2bf(float f) {
    unsigned u = __float_as_uint(f);
    u += 0x7FFFu + ((u >> 16) & 1u);   // round-to-nearest-even
    return (unsigned short)(u >> 16);
}

// ---------------------------------------------------------------------------
// fp32 -> bf16 bulk convert (vectorized float4 -> ushort4)
// ---------------------------------------------------------------------------
__global__ __launch_bounds__(256) void cvt_f32_bf16_kernel(
    const float4* __restrict__ in, ushort4* __restrict__ out, long long n4)
{
    long long i = (long long)blockIdx.x * 256 + threadIdx.x;
    if (i >= n4) return;
    float4 f = in[i];
    ushort4 o;
    o.x = f2bf(f.x); o.y = f2bf(f.y); o.z = f2bf(f.z); o.w = f2bf(f.w);
    out[i] = o;
}

// ---------------------------------------------------------------------------
// fp32 [R][C] -> bf16 [C][R] transpose-convert (32x32 tiles via LDS)
// ---------------------------------------------------------------------------
__global__ __launch_bounds__(256) void cvtT_f32_bf16_kernel(
    const float* __restrict__ in, unsigned short* __restrict__ out, int R, int C)
{
    __shared__ float tile[32][33];
    int tx = threadIdx.x & 31, ty = threadIdx.x >> 5;   // 32 x 8
    int c0 = blockIdx.x * 32, r0 = blockIdx.y * 32;
#pragma unroll
    for (int i = 0; i < 32; i += 8)
        tile[ty + i][tx] = in[(long long)(r0 + ty + i) * C + c0 + tx];
    __syncthreads();
#pragma unroll
    for (int i = 0; i < 32; i += 8)
        out[(long long)(c0 + ty + i) * R + r0 + tx] = f2bf(tile[tx][ty + i]);
}

// ---------------------------------------------------------------------------
// Row softmax: fp32 [rows][2048] -> bf16 [rows][2048]
// ---------------------------------------------------------------------------
__global__ __launch_bounds__(256) void softmax_rows_kernel(
    const float* __restrict__ S, unsigned short* __restrict__ P)
{
    __shared__ float red[256];
    const int tid = threadIdx.x;
    const long long row = blockIdx.x;
    const float* src = S + row * 2048;
    unsigned short* dst = P + row * 2048;

    float vals[8];
    float m = -3.402823466e38f;
#pragma unroll
    for (int i = 0; i < 8; ++i) {
        vals[i] = src[tid + i * 256];
        m = fmaxf(m, vals[i]);
    }
    red[tid] = m; __syncthreads();
    for (int s = 128; s > 0; s >>= 1) {
        if (tid < s) red[tid] = fmaxf(red[tid], red[tid + s]);
        __syncthreads();
    }
    m = red[0]; __syncthreads();

    float sum = 0.f;
#pragma unroll
    for (int i = 0; i < 8; ++i) { vals[i] = __expf(vals[i] - m); sum += vals[i]; }
    red[tid] = sum; __syncthreads();
    for (int s = 128; s > 0; s >>= 1) {
        if (tid < s) red[tid] += red[tid + s];
        __syncthreads();
    }
    float inv = 1.0f / red[0];
#pragma unroll
    for (int i = 0; i < 8; ++i) dst[tid + i * 256] = f2bf(vals[i] * inv);
}

// ---------------------------------------------------------------------------
// bf16 WMMA GEMM:  C[M,N] = alpha * (A[M,K] @ B) (+ bias[col])
//   A : bf16 row-major, leading dim lda
//   B : if B_NK : bf16 stored N x K row-major (i.e. B^T), ld = ldb
//       else    : bf16 stored K x N row-major, ld = ldb  (transposed via LDS)
//   C : fp32 or bf16 row-major, leading dim ldc
// Block tile 128x128, K-tile 64. 256 threads = 8 waves (2x4), wave tile 64x32.
// LDS ping-pong double buffering; next tile staged in 8 individually-named
// uint4 SSA scalars (macros, no lambdas/arrays -> no alloca -> no scratch).
// Requires: M%128==0, N%128==0, K%64==0 (true for all launches here).
// ---------------------------------------------------------------------------
#define GBM 128
#define GBN 128
#define GBK 64
#define NTHR 256
#define LSTR 72     // padded LDS row stride in bf16 elements (144B, 16B aligned)
#define LBUF (GBM * LSTR)   // one A (or B) buffer, in bf16 elements

// fetch one K-tile (A: 4 chunks, B: 4 chunks) into named registers
#define FETCH_TILE(k0)                                                         \
    do {                                                                       \
        ra0 = *(const uint4*)(&gAbase[(long long)rA0 * lda + (k0) + oA0]);     \
        ra1 = *(const uint4*)(&gAbase[(long long)rA1 * lda + (k0) + oA1]);     \
        ra2 = *(const uint4*)(&gAbase[(long long)rA2 * lda + (k0) + oA2]);     \
        ra3 = *(const uint4*)(&gAbase[(long long)rA3 * lda + (k0) + oA3]);     \
        if (B_NK) {                                                            \
            rb0 = *(const uint4*)(&gBbase[(long long)rA0 * ldb + (k0) + oA0]); \
            rb1 = *(const uint4*)(&gBbase[(long long)rA1 * ldb + (k0) + oA1]); \
            rb2 = *(const uint4*)(&gBbase[(long long)rA2 * ldb + (k0) + oA2]); \
            rb3 = *(const uint4*)(&gBbase[(long long)rA3 * ldb + (k0) + oA3]); \
        } else {                                                               \
            rb0 = *(const uint4*)(&gBbase[((k0) + kB0) * ldb + nB0]);          \
            rb1 = *(const uint4*)(&gBbase[((k0) + kB1) * ldb + nB1]);          \
            rb2 = *(const uint4*)(&gBbase[((k0) + kB2) * ldb + nB2]);          \
            rb3 = *(const uint4*)(&gBbase[((k0) + kB3) * ldb + nB3]);          \
        }                                                                      \
    } while (0)

// scatter-transpose one 16B chunk (8 bf16 along N) into LDS [N][K]
#define STB_CHUNK(v, kk, n0)                                                   \
    sB[((n0) + 0) * LSTR + (kk)] = (unsigned short)((v).x);                    \
    sB[((n0) + 1) * LSTR + (kk)] = (unsigned short)((v).x >> 16);              \
    sB[((n0) + 2) * LSTR + (kk)] = (unsigned short)((v).y);                    \
    sB[((n0) + 3) * LSTR + (kk)] = (unsigned short)((v).y >> 16);              \
    sB[((n0) + 4) * LSTR + (kk)] = (unsigned short)((v).z);                    \
    sB[((n0) + 5) * LSTR + (kk)] = (unsigned short)((v).z >> 16);              \
    sB[((n0) + 6) * LSTR + (kk)] = (unsigned short)((v).w);                    \
    sB[((n0) + 7) * LSTR + (kk)] = (unsigned short)((v).w >> 16);

// store staged registers into LDS buffer `buf`
#define STAGE_TILE(buf)                                                        \
    do {                                                                       \
        unsigned short* sA = As + (buf) * LBUF;                                \
        unsigned short* sB = Bs + (buf) * LBUF;                                \
        *(uint4*)(&sA[rA0 * LSTR + oA0]) = ra0;                                \
        *(uint4*)(&sA[rA1 * LSTR + oA1]) = ra1;                                \
        *(uint4*)(&sA[rA2 * LSTR + oA2]) = ra2;                                \
        *(uint4*)(&sA[rA3 * LSTR + oA3]) = ra3;                                \
        if (B_NK) {                                                            \
            *(uint4*)(&sB[rA0 * LSTR + oA0]) = rb0;                            \
            *(uint4*)(&sB[rA1 * LSTR + oA1]) = rb1;                            \
            *(uint4*)(&sB[rA2 * LSTR + oA2]) = rb2;                            \
            *(uint4*)(&sB[rA3 * LSTR + oA3]) = rb3;                            \
        } else {                                                               \
            STB_CHUNK(rb0, kB0, nB0)                                           \
            STB_CHUNK(rb1, kB1, nB1)                                           \
            STB_CHUNK(rb2, kB2, nB2)                                           \
            STB_CHUNK(rb3, kB3, nB3)                                           \
        }                                                                      \
    } while (0)

template<bool OUT_BF16, bool B_NK, bool HAS_BIAS>
__global__ __launch_bounds__(NTHR)
__attribute__((amdgpu_waves_per_eu(1)))
void gemm_bf16_wmma_kernel(
    const unsigned short* __restrict__ A, long long lda, long long strideA,
    const unsigned short* __restrict__ B, long long ldb, long long strideB,
    void* __restrict__ Cv, long long ldc, long long strideC,
    const float* __restrict__ bias, float alpha, int M, int N, int K)
{
    __shared__ unsigned short As[2 * LBUF];   // 2 x 18 KB
    __shared__ unsigned short Bs[2 * LBUF];   // 2 x 18 KB

    const int tid  = threadIdx.x;
    const int wave = tid >> 5;          // wave32: 0..7
    const int lane = tid & 31;
    const int wm = wave >> 2;           // 0..1  (M direction, 64 rows each)
    const int wn = wave & 3;            // 0..3  (N direction, 32 cols each)
    const int laneHi = lane >> 4;       // 0 or 1 (lane half)
    const int lane15 = lane & 15;

    const long long z = blockIdx.z;
    A += z * strideA;
    B += z * strideB;
    const long long rowBlock = (long long)blockIdx.y * GBM;
    const long long colBlock = (long long)blockIdx.x * GBN;

    const unsigned short* gAbase = A + rowBlock * lda;
    const unsigned short* gBbase = B_NK ? (B + colBlock * ldb) : (B + colBlock);

    // per-thread chunk coordinates (compile-time folded per chunk index)
    const int rA0 = (tid + 0 * NTHR) >> 3, oA0 = ((tid + 0 * NTHR) & 7) * 8;
    const int rA1 = (tid + 1 * NTHR) >> 3, oA1 = ((tid + 1 * NTHR) & 7) * 8;
    const int rA2 = (tid + 2 * NTHR) >> 3, oA2 = ((tid + 2 * NTHR) & 7) * 8;
    const int rA3 = (tid + 3 * NTHR) >> 3, oA3 = ((tid + 3 * NTHR) & 7) * 8;
    const int kB0 = (tid + 0 * NTHR) >> 4, nB0 = ((tid + 0 * NTHR) & 15) * 8;
    const int kB1 = (tid + 1 * NTHR) >> 4, nB1 = ((tid + 1 * NTHR) & 15) * 8;
    const int kB2 = (tid + 2 * NTHR) >> 4, nB2 = ((tid + 2 * NTHR) & 15) * 8;
    const int kB3 = (tid + 3 * NTHR) >> 4, nB3 = ((tid + 3 * NTHR) & 15) * 8;

    v8f acc[4][2];
#pragma unroll
    for (int mt = 0; mt < 4; ++mt)
#pragma unroll
        for (int nt = 0; nt < 2; ++nt)
            acc[mt][nt] = {};

    uint4 ra0, ra1, ra2, ra3, rb0, rb1, rb2, rb3;

    // ---- WMMA over LDS buffer `buf` (lambda: acc stays register-resident) -
    auto compute = [&](int buf) {
        const unsigned short* sA = As + buf * LBUF;
        const unsigned short* sB = Bs + buf * LBUF;
#pragma unroll
        for (int kk = 0; kk < GBK; kk += 32) {
            union Frag { v16bf v; v8bf h[2]; };
            Frag a[4], b[2];
            // A 16x32 fragment: lanes 0-15 -> K {0..7,16..23},
            //                   lanes 16-31 -> K {8..15,24..31}
#pragma unroll
            for (int mt = 0; mt < 4; ++mt) {
                int r  = wm * 64 + mt * 16 + lane15;
                int kb = kk + laneHi * 8;
                a[mt].h[0] = *(const v8bf*)(&sA[r * LSTR + kb]);
                a[mt].h[1] = *(const v8bf*)(&sA[r * LSTR + kb + 16]);
            }
            // B 32x16 fragment: lane holds col N, K {0..7,16..23} / {8..15,24..31} per half
#pragma unroll
            for (int nt = 0; nt < 2; ++nt) {
                int n  = wn * 32 + nt * 16 + lane15;
                int kb = kk + laneHi * 8;
                b[nt].h[0] = *(const v8bf*)(&sB[n * LSTR + kb]);
                b[nt].h[1] = *(const v8bf*)(&sB[n * LSTR + kb + 16]);
            }
#pragma unroll
            for (int mt = 0; mt < 4; ++mt)
#pragma unroll
                for (int nt = 0; nt < 2; ++nt)
                    acc[mt][nt] = __builtin_amdgcn_wmma_f32_16x16x32_bf16(
                        false, a[mt].v, false, b[nt].v,
                        (short)0, acc[mt][nt], false, false);
        }
    };

    // ---- pipelined main loop (ping-pong, one barrier per K-tile) ---------
    FETCH_TILE(0);
    STAGE_TILE(0);
    __syncthreads();
    int cur = 0;
    for (long long k0 = GBK; k0 < K; k0 += GBK) {
        FETCH_TILE(k0);     // global loads fly under the WMMA stream
        compute(cur);       // consume current LDS buffer
        STAGE_TILE(cur ^ 1);// wait loadcnt, fill the other buffer
        __syncthreads();
        cur ^= 1;
    }
    compute(cur);           // last tile

    // ---- epilogue: C layout VGPR e -> row (laneHi*8 + e), col lane15 ----
#pragma unroll
    for (int nt = 0; nt < 2; ++nt) {
        long long col = colBlock + wn * 32 + nt * 16 + lane15;
        float bv = HAS_BIAS ? bias[col] : 0.0f;
#pragma unroll
        for (int mt = 0; mt < 4; ++mt) {
            long long rbase = rowBlock + wm * 64 + mt * 16 + laneHi * 8;
#pragma unroll
            for (int e = 0; e < 8; ++e) {
                float v = acc[mt][nt][e] * alpha + bv;
                long long idx = z * strideC + (rbase + e) * ldc + col;
                if (OUT_BF16)
                    ((unsigned short*)Cv)[idx] = f2bf(v);
                else
                    ((float*)Cv)[idx] = v;
            }
        }
    }
}

// ---------------------------------------------------------------------------
// Host launcher
// ---------------------------------------------------------------------------
extern "C" void kernel_launch(void* const* d_in, const int* in_sizes, int n_in,
                              void* d_out, int out_size, void* d_ws, size_t ws_size,
                              hipStream_t stream)
{
    (void)in_sizes; (void)n_in; (void)out_size; (void)ws_size;

    const int Bsz = 4, S = 2048, H = 1024;
    const long long MS = (long long)Bsz * S;          // 8192

    const float* X     = (const float*)d_in[0];       // [4,2048,1024]
    const float* Wqkv  = (const float*)d_in[1];       // [1024,3072]
    const float* bqkv  = (const float*)d_in[2];       // [3072]
    const float* Wout  = (const float*)d_in[3];       // [1024,1024]
    const float* bout  = (const float*)d_in[4];       // [1024]
    float* out         = (float*)d_out;               // [4,2048,1024]

    // workspace partition (256B aligned regions)
    size_t off = 0;
    auto alloc = [&](size_t bytes) -> void* {
        void* p = (char*)d_ws + off;
        off += (bytes + 255) & ~(size_t)255;
        return p;
    };
    unsigned short* Xb    = (unsigned short*)alloc(MS * H * 2);            // 16MB
    unsigned short* WqkvT = (unsigned short*)alloc((size_t)3 * H * H * 2); //  6MB
    unsigned short* WoutT = (unsigned short*)alloc((size_t)H * H * 2);     //  2MB
    unsigned short* QKVb  = (unsigned short*)alloc(MS * 3 * H * 2);        // 48MB
    float*          S32   = (float*)alloc((size_t)Bsz * S * S * 4);        // 64MB
    unsigned short* P16   = (unsigned short*)alloc((size_t)Bsz * S * S * 2);// 32MB
    unsigned short* AO16  = (unsigned short*)alloc(MS * H * 2);            // 16MB

    dim3 blk(NTHR);

    // 1) convert inputs to bf16 (weights transposed so B is N x K rows)
    {
        long long n4 = MS * H / 4;
        cvt_f32_bf16_kernel<<<dim3((unsigned)((n4 + 255) / 256)), dim3(256), 0, stream>>>(
            (const float4*)X, (ushort4*)Xb, n4);
        cvtT_f32_bf16_kernel<<<dim3(3 * H / 32, H / 32), dim3(256), 0, stream>>>(
            Wqkv, WqkvT, H, 3 * H);
        cvtT_f32_bf16_kernel<<<dim3(H / 32, H / 32), dim3(256), 0, stream>>>(
            Wout, WoutT, H, H);
    }

    // 2) QKV projection: [8192,1024] @ [1024,3072] + b  -> bf16 [8192,3072]
    gemm_bf16_wmma_kernel<true, true, true>
        <<<dim3(3 * H / GBN, (unsigned)(MS / GBM), 1), blk, 0, stream>>>(
            Xb, H, 0, WqkvT, H, 0, QKVb, 3 * H, 0, bqkv, 1.0f,
            (int)MS, 3 * H, H);

    // 3) scores = Q @ K^T / sqrt(H)  (per batch) -> fp32 [4,2048,2048]
    gemm_bf16_wmma_kernel<false, true, false>
        <<<dim3(S / GBN, S / GBM, Bsz), blk, 0, stream>>>(
            QKVb,        3 * H, (long long)S * 3 * H,   // Q (cols 0..H-1)
            QKVb + H,    3 * H, (long long)S * 3 * H,   // K rows == B^T rows
            S32,         S,     (long long)S * S,
            nullptr, 0.03125f, S, S, H);

    // 4) softmax rows -> bf16 probabilities
    softmax_rows_kernel<<<dim3((unsigned)(Bsz * S)), dim3(256), 0, stream>>>(S32, P16);

    // 5) attn @ V (per batch): [2048,2048] @ [2048,1024] -> bf16 [4,2048,1024]
    gemm_bf16_wmma_kernel<true, false, false>
        <<<dim3(H / GBN, S / GBM, Bsz), blk, 0, stream>>>(
            P16,          S,     (long long)S * S,
            QKVb + 2 * H, 3 * H, (long long)S * 3 * H,  // V in K x N layout
            AO16,         H,     (long long)S * H,
            nullptr, 1.0f, S, H, S);

    // 6) output projection: [8192,1024] @ [1024,1024] + b -> fp32 out
    gemm_bf16_wmma_kernel<false, true, true>
        <<<dim3(H / GBN, (unsigned)(MS / GBM), 1), blk, 0, stream>>>(
            AO16, H, 0, WoutT, H, 0, out, H, 0, bout, 1.0f,
            (int)MS, H, H);
}
